// WanChannelLinearAttention_85564338471356
// MI455X (gfx1250) — compile-verified, matrix operand order checked by reading, not run on hardware
//
#include <hip/hip_runtime.h>
#include <hip/hip_bf16.h>

#define B_ 2
#define C_ 512
#define T_ 2048
#define D_ 64

typedef __attribute__((ext_vector_type(16))) _Float16 v16h;
typedef __attribute__((ext_vector_type(8)))  float    v8f;

__device__ __forceinline__ float elu1(float v) {
    // elu(v) + 1 == v+1 (v>0) else exp(v)
    return v > 0.f ? v + 1.f : __expf(v);
}

// ---------------------------------------------------------------------------
// Convert w_proj (f32, [C][C]) to f16 once; consumed 64x by the GEMM waves.
// ---------------------------------------------------------------------------
__global__ __launch_bounds__(256) void w_cvt(
    const float* __restrict__ W, _Float16* __restrict__ W16)
{
    // 8 floats per thread: 256 threads * 128 blocks * 8 = 262144 = C_*C_
    const int base = (blockIdx.x * 256 + threadIdx.x) * 8;
    const float4 f0 = *(const float4*)(W + base);
    const float4 f1 = *(const float4*)(W + base + 4);
    _Float16 h[8] = {
        (_Float16)f0.x, (_Float16)f0.y, (_Float16)f0.z, (_Float16)f0.w,
        (_Float16)f1.x, (_Float16)f1.y, (_Float16)f1.z, (_Float16)f1.w };
    *(float4*)(W16 + base) = *(const float4*)h;
}

// ---------------------------------------------------------------------------
// Stage 1: per-(b,t) channel linear attention. One block per (b,t).
// Writes attn as f16 in [B][T][C] layout (c contiguous) for WMMA B-fragments.
// ---------------------------------------------------------------------------
__global__ __launch_bounds__(256) void attn_stage(
    const float* __restrict__ x,   // [B][C][T]
    const float* __restrict__ wq, const float* __restrict__ bq,
    const float* __restrict__ wk, const float* __restrict__ bk,
    const float* __restrict__ wv, const float* __restrict__ bv,
    _Float16* __restrict__ attn)   // [B][T][C]
{
    __shared__ float xs[C_];
    __shared__ float Sp[4][D_];
    __shared__ float Zp[4][D_];
    __shared__ float Sv[D_], Zv[D_], wqs[D_], bqs[D_];

    const int bt  = blockIdx.x;       // b*T + t
    const int b   = bt / T_;
    const int t   = bt - b * T_;
    const int tid = threadIdx.x;

    const float wv0 = wv[0], bv0 = bv[0];

    // stage x[b, :, t] into LDS (strided in global; L2-resident)
    for (int c = tid; c < C_; c += 256)
        xs[c] = x[((size_t)b * C_ + c) * T_ + t];
    if (tid < D_) { wqs[tid] = wq[tid]; bqs[tid] = bq[tid]; }
    __syncthreads();

    // Phase B: S[d] = sum_c phi_k(c,d)*v(c), Z[d] = sum_c phi_k(c,d)
    // thread = (d, group-of-128-channels)
    const int d = tid & 63;
    const int g = tid >> 6;
    const float wkd = wk[d], bkd = bk[d];
    float s = 0.f, z = 0.f;
    const int c0 = g * 128;
#pragma unroll 4
    for (int c = c0; c < c0 + 128; ++c) {
        const float xc = xs[c];
        const float ph = elu1(fmaf(xc, wkd, bkd));
        z += ph;
        s += ph * fmaf(xc, wv0, bv0);
    }
    Sp[g][d] = s; Zp[g][d] = z;
    __syncthreads();
    if (tid < D_) {
        Sv[tid] = Sp[0][tid] + Sp[1][tid] + Sp[2][tid] + Sp[3][tid];
        Zv[tid] = fmaxf(Zp[0][tid] + Zp[1][tid] + Zp[2][tid] + Zp[3][tid], 1e-6f);
    }
    __syncthreads();

    // Phase C: per-channel numerator/denominator over d
#pragma unroll
    for (int j = 0; j < 2; ++j) {
        const int c = tid + j * 256;
        const float xc = xs[c];
        float num = 0.f, den = 0.f;
#pragma unroll 8
        for (int dd = 0; dd < D_; ++dd) {
            const float ph = elu1(fmaf(xc, wqs[dd], bqs[dd]));
            num = fmaf(ph, Sv[dd], num);
            den = fmaf(ph, Zv[dd], den);
        }
        attn[(size_t)bt * C_ + c] = (_Float16)(num / den);
    }
}

// ---------------------------------------------------------------------------
// Stage 2: out[b,o,t] = x[b,o,t] + b_proj[o] + sum_c W[o,c]*attn[b,t,c]
// WMMA f32 <- f16 16x16x32. One wave computes a 16(M) x 64(N=t) strip.
// ---------------------------------------------------------------------------
union FragH { v16h h; float4 f[2]; };

__global__ __launch_bounds__(256) void proj_gemm(
    const float*    __restrict__ x,     // [B][C][T]
    const _Float16* __restrict__ attn,  // [B][T][C]
    const _Float16* __restrict__ W16,   // [C][C] row-major (o, c), f16
    const float*    __restrict__ bproj, // [C]
    float*          __restrict__ out)   // [B][C][T]
{
    const int wave = (blockIdx.x * blockDim.x + threadIdx.x) >> 5;
    const int lane = threadIdx.x & 31;

    const int b   = wave >> 10;          // 1024 waves per batch
    const int rem = wave & 1023;
    const int m0  = (rem >> 5) << 4;     // 32 M-tiles of 16
    const int t0  = (rem & 31) << 6;     // 32 N-strips of 64

    const int lm = lane & 15;            // row (A) / col (B) within tile
    const int kh = (lane >> 4) << 3;     // K half-select: 0 or 8 (ISA 16-bit layout)

    const _Float16* wrow  = W16 + (size_t)(m0 + lm) * C_;
    const _Float16* abase = attn + (size_t)b * T_ * C_;

    v8f acc0 = {}, acc1 = {}, acc2 = {}, acc3 = {};

    for (int kb = 0; kb < C_; kb += 32) {
        // --- A fragment: W16 rows.
        // lane<16: K = {kb+0..7, kb+16..23}; lane>=16: K = {kb+8..15, kb+24..31}
        FragH af;
        {
            const _Float16* wp = wrow + kb + kh;
            af.f[0] = *(const float4*)(wp);
            af.f[1] = *(const float4*)(wp + 16);
        }

        // --- B fragments: attn[t][c], c contiguous -> two 16B loads per lane
        FragH bf0, bf1, bf2, bf3;
        {
            const _Float16* ap = abase + (size_t)(t0 + 0  + lm) * C_ + kb + kh;
            bf0.f[0] = *(const float4*)(ap);
            bf0.f[1] = *(const float4*)(ap + 16);
        }
        {
            const _Float16* ap = abase + (size_t)(t0 + 16 + lm) * C_ + kb + kh;
            bf1.f[0] = *(const float4*)(ap);
            bf1.f[1] = *(const float4*)(ap + 16);
        }
        {
            const _Float16* ap = abase + (size_t)(t0 + 32 + lm) * C_ + kb + kh;
            bf2.f[0] = *(const float4*)(ap);
            bf2.f[1] = *(const float4*)(ap + 16);
        }
        {
            const _Float16* ap = abase + (size_t)(t0 + 48 + lm) * C_ + kb + kh;
            bf3.f[0] = *(const float4*)(ap);
            bf3.f[1] = *(const float4*)(ap + 16);
        }

        acc0 = __builtin_amdgcn_wmma_f32_16x16x32_f16(false, af.h, false, bf0.h,
                                                      (short)0, acc0, false, false);
        acc1 = __builtin_amdgcn_wmma_f32_16x16x32_f16(false, af.h, false, bf1.h,
                                                      (short)0, acc1, false, false);
        acc2 = __builtin_amdgcn_wmma_f32_16x16x32_f16(false, af.h, false, bf2.h,
                                                      (short)0, acc2, false, false);
        acc3 = __builtin_amdgcn_wmma_f32_16x16x32_f16(false, af.h, false, bf3.h,
                                                      (short)0, acc3, false, false);
    }

    // Epilogue: D layout = VGPR r: lanes 0-15 -> M=r, lanes 16-31 -> M=r+8, N=lane&15
    const int mtop = (lane >> 4) << 3;   // 0 or 8
    auto store_tile = [&](const v8f& acc, int j) {
        const int tcol = t0 + j * 16 + lm;
#pragma unroll
        for (int r = 0; r < 8; ++r) {
            const int m = m0 + mtop + r;
            const size_t idx = ((size_t)b * C_ + m) * T_ + tcol;
            out[idx] = x[idx] + bproj[m] + acc[r];
        }
    };
    store_tile(acc0, 0);
    store_tile(acc1, 1);
    store_tile(acc2, 2);
    store_tile(acc3, 3);
}

extern "C" void kernel_launch(void* const* d_in, const int* in_sizes, int n_in,
                              void* d_out, int out_size, void* d_ws, size_t ws_size,
                              hipStream_t stream) {
    const float* x      = (const float*)d_in[0];
    const float* wq     = (const float*)d_in[1];
    const float* bq     = (const float*)d_in[2];
    const float* wk     = (const float*)d_in[3];
    const float* bk     = (const float*)d_in[4];
    const float* wv     = (const float*)d_in[5];
    const float* bv     = (const float*)d_in[6];
    const float* w_proj = (const float*)d_in[7];
    const float* b_proj = (const float*)d_in[8];

    // ws layout: [0, 4MB): attn f16 [B][T][C];  [4MB, 4.5MB): w_proj f16
    _Float16* attn = (_Float16*)d_ws;
    _Float16* W16  = (_Float16*)((char*)d_ws + (size_t)B_ * T_ * C_ * sizeof(_Float16));

    w_cvt<<<(C_ * C_) / (256 * 8), 256, 0, stream>>>(w_proj, W16);
    attn_stage<<<B_ * T_, 256, 0, stream>>>(x, wq, bq, wk, bk, wv, bv, attn);

    // 2048 waves total: B(2) * Mtiles(32) * Nstrips(32); 8 waves/block
    proj_gemm<<<256, 256, 0, stream>>>(x, attn, W16, b_proj, (float*)d_out);
}
